// QuantLinear_16020228014487
// MI455X (gfx1250) — compile-verified, matrix operand order checked by reading, not run on hardware
//
#include <hip/hip_runtime.h>
#include <hip/hip_bf16.h>

// ---------------------------------------------------------------------------
// QuantLinear + COFT forward for MI455X (gfx1250, wave32).
//   out = (x @ blockdiag(Q)) @ W^T + bias
//   Q = Cayley(project(oft_R)) = I - 2S + O(S^2), ||S||_F <= 2.5e-6  -> Q = I-2S
//   W = (qweight-128)*scales*0.01, dequantized once to bf16.
// Main GEMM: V_WMMA_F32_16X16X32_BF16, LDS double-buffered, tiles moved by
// GLOBAL_LOAD_ASYNC_TO_LDS_B128 (ASYNCcnt) + TENSOR_LOAD_TO_LDS (TDM,
// TENSORcnt), with synchronous fallbacks if the builtins are absent.
// ---------------------------------------------------------------------------

typedef __attribute__((ext_vector_type(16))) __bf16    v16bf;
typedef __attribute__((ext_vector_type(8)))  __bf16    v8bf;
typedef __attribute__((ext_vector_type(4)))  __bf16    v4bf;
typedef __attribute__((ext_vector_type(8)))  float     v8f;
typedef __attribute__((ext_vector_type(4)))  unsigned  v4u;
typedef __attribute__((ext_vector_type(8)))  int       v8i;
typedef __attribute__((ext_vector_type(4)))  int       v4i;

// address-space-qualified v4i for the async-to-LDS builtin's parameters
typedef __attribute__((address_space(1))) v4i GAS_v4i;   // global (__device__)
typedef __attribute__((address_space(3))) v4i LAS_v4i;   // LDS (__shared__)

#define BS_TOT 4096
#define IN_F   4096
#define OUT_F  4096
#define RBLK   16
#define BLK    256
#define EPS_B  (1e-5f / 4.0f)

#define HAS_ASYNC __has_builtin(__builtin_amdgcn_global_load_async_to_lds_b128)
#define HAS_WAITA __has_builtin(__builtin_amdgcn_s_wait_asynccnt)
#define HAS_TDM   __has_builtin(__builtin_amdgcn_tensor_load_to_lds)
#define HAS_WAITT __has_builtin(__builtin_amdgcn_s_wait_tensorcnt)

static __device__ inline v16bf pack16(v8bf lo, v8bf hi) {
  return __builtin_shufflevector(lo, hi, 0,1,2,3,4,5,6,7,8,9,10,11,12,13,14,15);
}

// 16-byte global->LDS copy: async (ASYNCcnt) when available, else direct.
static __device__ inline void cp16_g2l(const __bf16* g, __bf16* l) {
#if HAS_ASYNC
  __builtin_amdgcn_global_load_async_to_lds_b128(
      (GAS_v4i*)(v4i*)(__bf16*)g,
      (LAS_v4i*)(v4i*)l, 0, 0);
#else
  *(v8bf*)l = *(const v8bf*)g;
#endif
}

static __device__ inline unsigned lds_byte_addr(const void* p) {
  return (unsigned)(unsigned long long)
      (__attribute__((address_space(3))) const void*)p;
}

// TDM: load a 128(row) x 32(col) bf16 tile (row stride IN_F) into LDS.
// Descriptor per ISA 8.3/8.4: 2D tensor, data_size=2B, compact LDS tile.
static __device__ inline void tdm_load_tile_128x32(const __bf16* gsrc,
                                                   unsigned lds_byte) {
#if HAS_TDM
  const unsigned long long ga = (unsigned long long)gsrc;
  v4u g0;
  g0[0] = 1u;                                             // count=1 (valid)
  g0[1] = lds_byte;                                       // lds_addr
  g0[2] = (unsigned)ga;                                   // global_addr[31:0]
  g0[3] = (unsigned)((ga >> 32) & 0x01FFFFFFu) | (2u << 30); // [56:32] | type=2
  const unsigned td0 = IN_F, td1 = OUT_F;                 // tensor dims
  const unsigned tl0 = 32u,  tl1 = 128u;                  // tile dims
  const unsigned s0  = IN_F;                              // dim0 stride (elems)
  v8i g1;
  g1[0] = (int)(1u << 16);                                // data_size=1 -> 2B
  g1[1] = (int)((td0 & 0xFFFFu) << 16);                   // tensor_dim0 lo16
  g1[2] = (int)((td0 >> 16) | ((td1 & 0xFFFFu) << 16));   // d0 hi | d1 lo
  g1[3] = (int)((td1 >> 16) | (tl0 << 16));               // d1 hi | tile_dim0
  g1[4] = (int)tl1;                                       // tile_dim1, dim2=0
  g1[5] = (int)s0;                                        // dim0_stride lo32
  g1[6] = 0;                                              // stride hi | d1s lo
  g1[7] = 0;
  const v4i z4 = {0, 0, 0, 0};
#if __clang_major__ >= 23
  const v8i z8 = {0, 0, 0, 0, 0, 0, 0, 0};
  __builtin_amdgcn_tensor_load_to_lds(g0, g1, z4, z4, z8, 0);
#else
  __builtin_amdgcn_tensor_load_to_lds(g0, g1, z4, z4, 0);
#endif
#endif
}

// --- K1: per-block Frobenius norm of oft_R ---------------------------------
__global__ void oft_norm_kernel(const float* __restrict__ R, float* __restrict__ norms) {
  __shared__ float red[256];
  const int r = blockIdx.x;
  const float* p = R + (r << 16);
  float s = 0.f;
  for (int i = threadIdx.x; i < BLK * BLK; i += 256) { float v = p[i]; s += v * v; }
  red[threadIdx.x] = s;
  __syncthreads();
  for (int off = 128; off; off >>= 1) {
    if (threadIdx.x < off) red[threadIdx.x] += red[threadIdx.x + off];
    __syncthreads();
  }
  if (threadIdx.x == 0) norms[r] = sqrtf(red[0]);
}

// --- K2: Bm[r][n][k] = E[r][k][n] = scale_r*(R[r][n][k]-R[r][k][n]) --------
__global__ void oft_build_e_kernel(const float* __restrict__ R,
                                   const float* __restrict__ norms,
                                   __bf16* __restrict__ Bm) {
  const int idx = blockIdx.x * 256 + threadIdx.x;
  const int r = idx >> 16, n = (idx >> 8) & 255, k = idx & 255;
  const float nr = norms[r];
  const float scale = (nr <= EPS_B) ? 1.0f : (EPS_B / nr);
  const float* Rr = R + (r << 16);
  Bm[idx] = (__bf16)(scale * (Rr[(n << 8) + k] - Rr[(k << 8) + n]));
}

// --- K3: dequantize weight to bf16 -----------------------------------------
__global__ void dequant_kernel(const int* __restrict__ q,
                               const float* __restrict__ scales,
                               __bf16* __restrict__ Wb) {
  const int idx = blockIdx.x * 256 + threadIdx.x;
  const int o = idx >> 10, k4 = (idx & 1023) << 2;
  const float sc = scales[o] * 0.01f;
  const int4 qi = *(const int4*)(q + (size_t)o * IN_F + k4);
  v4bf w;
  w[0] = (__bf16)(((float)qi.x - 128.f) * sc);
  w[1] = (__bf16)(((float)qi.y - 128.f) * sc);
  w[2] = (__bf16)(((float)qi.z - 128.f) * sc);
  w[3] = (__bf16)(((float)qi.w - 128.f) * sc);
  *(v4bf*)(Wb + (size_t)o * IN_F + k4) = w;
}

// --- K4: xr = x + x @ blockdiag(E), bf16 WMMA ------------------------------
__global__ void rotate_kernel(const float* __restrict__ x,
                              const __bf16* __restrict__ Bm,
                              __bf16* __restrict__ xr) {
  const int wave = threadIdx.x >> 5, lane = threadIdx.x & 31;
  const int half = lane >> 4, l16 = lane & 15;
  const int r = blockIdx.y >> 1, nh = blockIdx.y & 1;
  const int mbase = blockIdx.x * 64 + (wave >> 1) * 16;
  const int nbase = nh * 128 + (wave & 1) * 64;
  const __bf16* Bblk = Bm + (r << 16);
  const float*  xrow = x + (size_t)(mbase + l16) * IN_F + (r << 8);

  v8f acc[4] = {};
  for (int kt = 0; kt < BLK; kt += 32) {
    const int k0 = kt + half * 8;
    v16bf a;
#pragma unroll
    for (int t = 0; t < 8; ++t) a[t]     = (__bf16)xrow[k0 + t];
#pragma unroll
    for (int t = 0; t < 8; ++t) a[8 + t] = (__bf16)xrow[k0 + 16 + t];
#pragma unroll
    for (int t = 0; t < 4; ++t) {
      const __bf16* bp = Bblk + (size_t)(nbase + t * 16 + l16) * BLK + k0;
      const v16bf b = pack16(*(const v8bf*)bp, *(const v8bf*)(bp + 16));
      acc[t] = __builtin_amdgcn_wmma_f32_16x16x32_bf16(
          false, a, false, b, (short)0, acc[t], false, false);
    }
  }
#pragma unroll
  for (int t = 0; t < 4; ++t)
#pragma unroll
    for (int v = 0; v < 8; ++v) {
      const int m = mbase + v + 8 * half;
      const int n = (r << 8) + nbase + t * 16 + l16;
      xr[(size_t)m * IN_F + n] = (__bf16)(x[(size_t)m * IN_F + n] + acc[t][v]);
    }
}

// --- K5: out = xr @ Wb^T + bias --------------------------------------------
// WG tile 128x128, 8 waves (4M x 2N), wave tile 32x64 -> 8 WMMA per K-step.
// Double-buffered LDS; A tile via async-to-LDS, B tile via TDM (wave 0).
__global__ void gemm_kernel(const __bf16* __restrict__ xr,
                            const __bf16* __restrict__ Wb,
                            const float* __restrict__ bias,
                            float* __restrict__ out) {
  __shared__ __bf16 At[2][128][32];
  __shared__ __bf16 Bt[2][128][32];
  const int tid  = threadIdx.x;
  const int wave = tid >> 5, lane = tid & 31;
  const int half = lane >> 4, l16 = lane & 15;
  const int wm = wave >> 1, wn = wave & 1;       // 4 x 2 wave grid
  const int mblk = blockIdx.x * 128, nblk = blockIdx.y * 128;

  // per-thread staging chunks: 512 x 16B per tile, 2 chunks/thread
  const int crow = tid >> 2, ccol = (tid & 3) * 8;
  const __bf16* agp = xr + (size_t)(mblk + crow) * IN_F + ccol;
  const __bf16* bgp = Wb + (size_t)(nblk + crow) * IN_F + ccol;

  auto load_a = [&](int buf, int kt) {
    cp16_g2l(agp + kt, &At[buf][crow][ccol]);
    cp16_g2l(agp + kt + (size_t)64 * IN_F, &At[buf][crow + 64][ccol]);
  };
  auto load_b = [&](int buf, int kt) {
#if HAS_TDM
    if (wave == 0)
      tdm_load_tile_128x32(Wb + (size_t)nblk * IN_F + kt,
                           lds_byte_addr(&Bt[buf][0][0]));
#else
    cp16_g2l(bgp + kt, &Bt[buf][crow][ccol]);
    cp16_g2l(bgp + kt + (size_t)64 * IN_F, &Bt[buf][crow + 64][ccol]);
#endif
  };
  auto pipe_wait = [&]() {
#if HAS_ASYNC && HAS_WAITA
    __builtin_amdgcn_s_wait_asynccnt(0);
#endif
#if HAS_TDM && HAS_WAITT
    if (wave == 0) __builtin_amdgcn_s_wait_tensorcnt(0);
#endif
    __syncthreads();
  };

  v8f acc[2][4] = {};
  load_a(0, 0);
  load_b(0, 0);
  for (int kt = 0; kt < IN_F; kt += 32) {
    const int buf = (kt >> 5) & 1;
    pipe_wait();                                  // tiles[buf] ready WG-wide
    if (kt + 32 < IN_F) { load_a(buf ^ 1, kt + 32); load_b(buf ^ 1, kt + 32); }

    const int k0 = half * 8;
    const v16bf a0 = pack16(*(const v8bf*)&At[buf][wm * 32 + l16][k0],
                            *(const v8bf*)&At[buf][wm * 32 + l16][k0 + 16]);
    const v16bf a1 = pack16(*(const v8bf*)&At[buf][wm * 32 + 16 + l16][k0],
                            *(const v8bf*)&At[buf][wm * 32 + 16 + l16][k0 + 16]);
#pragma unroll
    for (int t = 0; t < 4; ++t) {
      const v16bf b = pack16(*(const v8bf*)&Bt[buf][wn * 64 + t * 16 + l16][k0],
                             *(const v8bf*)&Bt[buf][wn * 64 + t * 16 + l16][k0 + 16]);
      acc[0][t] = __builtin_amdgcn_wmma_f32_16x16x32_bf16(
          false, a0, false, b, (short)0, acc[0][t], false, false);
      acc[1][t] = __builtin_amdgcn_wmma_f32_16x16x32_bf16(
          false, a1, false, b, (short)0, acc[1][t], false, false);
    }
  }
#pragma unroll
  for (int f = 0; f < 2; ++f)
#pragma unroll
    for (int t = 0; t < 4; ++t)
#pragma unroll
      for (int v = 0; v < 8; ++v) {
        const int m = mblk + wm * 32 + f * 16 + v + 8 * half;
        const int n = nblk + wn * 64 + t * 16 + l16;
        out[(size_t)m * OUT_F + n] = acc[f][t][v] + bias[n];
      }
}

// ---------------------------------------------------------------------------
extern "C" void kernel_launch(void* const* d_in, const int* in_sizes, int n_in,
                              void* d_out, int out_size, void* d_ws, size_t ws_size,
                              hipStream_t stream) {
  const float* x      = (const float*)d_in[0];
  const int*   qw     = (const int*)d_in[1];
  const float* scales = (const float*)d_in[2];
  const float* bias   = (const float*)d_in[3];
  const float* oftR   = (const float*)d_in[4];
  float* out = (float*)d_out;

  char* ws = (char*)d_ws;
  float*  norms = (float*)ws;                                    // 64 B
  __bf16* Bm = (__bf16*)(ws + 1024);                             // 2 MB
  __bf16* Wb = (__bf16*)(ws + 1024 + (size_t)2 * 1024 * 1024);   // 32 MB
  __bf16* xr = (__bf16*)(ws + 1024 + (size_t)34 * 1024 * 1024);  // 32 MB

  oft_norm_kernel   <<<RBLK, 256, 0, stream>>>(oftR, norms);
  oft_build_e_kernel<<<(RBLK * BLK * BLK) / 256, 256, 0, stream>>>(oftR, norms, Bm);
  dequant_kernel    <<<(OUT_F * IN_F / 4) / 256, 256, 0, stream>>>(qw, scales, Wb);
  rotate_kernel     <<<dim3(BS_TOT / 64, RBLK * 2), 256, 0, stream>>>(x, Bm, xr);
  gemm_kernel       <<<dim3(BS_TOT / 128, OUT_F / 128), 256, 0, stream>>>(xr, Wb, bias, out);
}
